// NIC_Decoder_37194416783806
// MI455X (gfx1250) — matrix operand order repaired and checked by reading, not compiled, not run on hardware
//
#include <hip/hip_runtime.h>
#include <cstdint>
#include <cstddef>

// ---------------------------------------------------------------------------
// NIC LSTM decoder for gfx1250 (MI455X): bf16 WMMA (v_wmma_f32_16x16x32_bf16)
// for all GEMMs, f32 accumulate. Sequential recurrence = 64 kernel launches.
// ---------------------------------------------------------------------------

typedef __bf16 v16bf __attribute__((ext_vector_type(16)));
typedef __bf16 v8bf  __attribute__((ext_vector_type(8)));
typedef float  v8f   __attribute__((ext_vector_type(8)));

#define Bdim 32
#define Ldim 64
#define Edim 512
#define Udim 512
#define Vdim 32000
#define FourU 2048

#define KEEP 0.7f
#define INV_KEEP (1.0f / 0.7f)

__device__ __forceinline__ uint32_t hash_u32(uint32_t x) {
  x ^= x >> 16; x *= 0x7feb352dU;
  x ^= x >> 15; x *= 0x846ca68bU;
  x ^= x >> 16;
  return x;
}

// Deterministic bernoulli dropout mask value: 0 or 1/keep.
__device__ __forceinline__ float dropmask(uint32_t a, uint32_t b, uint32_t salt) {
  uint32_t h = hash_u32(a * 0x9E3779B9u + hash_u32(b * 0x85EBCA6Bu + salt));
  float r = (float)(h >> 8) * (1.0f / 16777216.0f);
  return r < KEEP ? INV_KEEP : 0.0f;
}

__device__ __forceinline__ float sigm(float x) { return 1.0f / (1.0f + __expf(-x)); }

__device__ __forceinline__ v16bf cat16(v8bf lo, v8bf hi) {
  return __builtin_shufflevector(lo, hi, 0,1,2,3,4,5,6,7,8,9,10,11,12,13,14,15);
}

// ---------------------------------------------------------------------------
// Prep kernels
// ---------------------------------------------------------------------------

// X[b*64+t][e] (bf16): t==0 -> images_emb; t>0 -> emb_table[targets[b][t-1]]*word_mask(b,e)
__global__ void k_build_x(const float* __restrict__ img,
                          const int*   __restrict__ targets,
                          const float* __restrict__ emb,
                          __bf16*      __restrict__ X) {
  int i = blockIdx.x * blockDim.x + threadIdx.x;      // [0, B*L*E)
  int e  = i & (Edim - 1);
  int bt = i >> 9;                                    // E = 512
  int t  = bt & (Ldim - 1);
  int b  = bt >> 6;
  float v;
  if (t == 0) {
    v = img[b * Edim + e];
  } else {
    int idx = targets[b * Ldim + (t - 1)];
    v = emb[(size_t)idx * Edim + e] * dropmask((uint32_t)b, (uint32_t)e, 1u);
  }
  X[i] = (__bf16)v;
}

// out[c][r] = bf16(in[r][c])   (weight transpose + downconvert)
__global__ void k_cvt_transpose(const float* __restrict__ in,
                                __bf16* __restrict__ out, int R, int C) {
  size_t i = (size_t)blockIdx.x * blockDim.x + threadIdx.x;
  if (i >= (size_t)R * C) return;
  int r = (int)(i / C);
  int c = (int)(i % C);
  out[(size_t)c * R + r] = (__bf16)in[i];
}

__global__ void k_init_state(float* __restrict__ H0, float* __restrict__ Cst) {
  int i = blockIdx.x * blockDim.x + threadIdx.x;
  if (i < Bdim * Udim) { H0[i] = 0.0f; Cst[i] = 0.0f; }
}

// ---------------------------------------------------------------------------
// Generic bf16 WMMA GEMM:  C[M][N] f32 = A[M][K] @ Bt[N][K]^T + bias[N]
// Each wave: one 16-row M tile x four 16-col N tiles (A-fragment reuse).
// ---------------------------------------------------------------------------
__global__ void k_gemm_bf16(const __bf16* __restrict__ A,
                            const __bf16* __restrict__ Bt,
                            const float*  __restrict__ bias,
                            float* __restrict__ C,
                            int M, int N, int K) {
  const int wave = blockIdx.x * (blockDim.x >> 5) + (threadIdx.x >> 5);
  const int lane = threadIdx.x & 31;
  const int ngroups = N >> 6;                      // groups of 4 N-tiles
  const int m_tile  = wave / ngroups;
  const int n_group = wave - m_tile * ngroups;
  if (m_tile * 16 >= M) return;

  const int l15  = lane & 15;
  const int half = lane >> 4;
  const int n_base = n_group << 6;

  // A fragment: lane holds row (lane&15); K chunks {kb..kb+7, kb+16..kb+23}, kb=(lane>>4)*8
  const __bf16* Ap  = A  + (size_t)(m_tile * 16 + l15) * K + half * 8;
  // B fragment: lane holds col (lane&15); K chunks {kb..kb+7, kb+16..kb+23}, kb=(lane>>4)*8
  const __bf16* Bp0 = Bt + (size_t)(n_base +  0 + l15) * K + half * 8;
  const __bf16* Bp1 = Bt + (size_t)(n_base + 16 + l15) * K + half * 8;
  const __bf16* Bp2 = Bt + (size_t)(n_base + 32 + l15) * K + half * 8;
  const __bf16* Bp3 = Bt + (size_t)(n_base + 48 + l15) * K + half * 8;

  v8f acc0 = {}, acc1 = {}, acc2 = {}, acc3 = {};
  for (int k0 = 0; k0 < K; k0 += 32) {
    v16bf a  = cat16(*(const v8bf*)(Ap  + k0), *(const v8bf*)(Ap  + k0 + 16));
    v16bf b0 = cat16(*(const v8bf*)(Bp0 + k0), *(const v8bf*)(Bp0 + k0 + 16));
    v16bf b1 = cat16(*(const v8bf*)(Bp1 + k0), *(const v8bf*)(Bp1 + k0 + 16));
    v16bf b2 = cat16(*(const v8bf*)(Bp2 + k0), *(const v8bf*)(Bp2 + k0 + 16));
    v16bf b3 = cat16(*(const v8bf*)(Bp3 + k0), *(const v8bf*)(Bp3 + k0 + 16));
    acc0 = __builtin_amdgcn_wmma_f32_16x16x32_bf16(false, a, false, b0, (short)0, acc0, false, false);
    acc1 = __builtin_amdgcn_wmma_f32_16x16x32_bf16(false, a, false, b1, (short)0, acc1, false, false);
    acc2 = __builtin_amdgcn_wmma_f32_16x16x32_bf16(false, a, false, b2, (short)0, acc2, false, false);
    acc3 = __builtin_amdgcn_wmma_f32_16x16x32_bf16(false, a, false, b3, (short)0, acc3, false, false);
  }

  const float bi0 = bias ? bias[n_base +  0 + l15] : 0.0f;
  const float bi1 = bias ? bias[n_base + 16 + l15] : 0.0f;
  const float bi2 = bias ? bias[n_base + 32 + l15] : 0.0f;
  const float bi3 = bias ? bias[n_base + 48 + l15] : 0.0f;

#pragma unroll
  for (int r = 0; r < 8; r++) {
    // C/D layout: VGPR r -> row (r + half*8), col (lane&15)
    const int m = m_tile * 16 + half * 8 + r;
    float* Crow = C + (size_t)m * N + n_base + l15;
    Crow[ 0] = acc0[r] + bi0;
    Crow[16] = acc1[r] + bi1;
    Crow[32] = acc2[r] + bi2;
    Crow[48] = acc3[r] + bi3;
  }
}

// ---------------------------------------------------------------------------
// One LSTM timestep. Grid = 32 blocks (16-wide u-slices), 256 threads (8 waves).
// Wave w: gate (w&3), batch-half (w>>2). Recurrent-dropout applied to h on the
// fly while building the A fragment; z_in pulled from precomputed Zin.
// ---------------------------------------------------------------------------
__global__ void k_lstm_step(const float*  __restrict__ Zin,    // [B*L][4U], row = b*64+t
                            const __bf16* __restrict__ WrecT,  // [4U][U]: row g*U+v over u
                            const float*  __restrict__ Hin,    // [B][U]
                            float*        __restrict__ Hout,   // [B][U]
                            float*        __restrict__ Cst,    // [B][U] (in/out, disjoint slices)
                            __bf16*       __restrict__ Hseq,   // [B*L][U] bf16 (fin-masked)
                            int t) {
  __shared__ float zsh[4][Bdim][16];

  const int u_slice = blockIdx.x * 16;
  const int wv   = threadIdx.x >> 5;
  const int lane = threadIdx.x & 31;
  const int g     = wv & 3;
  const int mhalf = wv >> 2;
  const int l15  = lane & 15;
  const int half = lane >> 4;

  const int b_row = mhalf * 16 + l15;                 // batch row for A fragment
  const float*  hrow = Hin + b_row * Udim;
  const __bf16* Brow = WrecT + (size_t)(g * Udim + u_slice + l15) * Udim + half * 8;

  v8f acc = {};
  for (int k0 = 0; k0 < Udim; k0 += 32) {
    const int ka = k0 + half * 8;
    v16bf a;
#pragma unroll
    for (int e = 0; e < 8; e++) {
      const int k1 = ka + e;
      const int k2 = ka + 16 + e;
      a[e]     = (__bf16)(hrow[k1] * dropmask((uint32_t)((g << 8) | b_row), (uint32_t)k1, 2u));
      a[e + 8] = (__bf16)(hrow[k2] * dropmask((uint32_t)((g << 8) | b_row), (uint32_t)k2, 2u));
    }
    v16bf b = cat16(*(const v8bf*)(Brow + k0), *(const v8bf*)(Brow + k0 + 16));
    acc = __builtin_amdgcn_wmma_f32_16x16x32_bf16(false, a, false, b, (short)0, acc, false, false);
  }

#pragma unroll
  for (int r = 0; r < 8; r++) {
    const int bb = mhalf * 16 + half * 8 + r;         // batch index of this C row
    const float z = acc[r] +
        Zin[(size_t)(bb * Ldim + t) * FourU + g * Udim + u_slice + l15];
    zsh[g][bb][l15] = z;
  }
  __syncthreads();

  // Pointwise LSTM update: 512 (b,u) pairs, 2 per thread.
  for (int p = threadIdx.x; p < Bdim * 16; p += blockDim.x) {
    const int b = p >> 4;
    const int u = p & 15;
    const int uidx = u_slice + u;
    const float zi = zsh[0][b][u], zf = zsh[1][b][u];
    const float zg = zsh[2][b][u], zo = zsh[3][b][u];
    const float c_old = Cst[b * Udim + uidx];
    const float i = sigm(zi);
    const float f = sigm(zf);
    const float gg = tanhf(zg);
    const float o = sigm(zo);
    const float c_new = f * c_old + i * gg;
    const float h_new = o * tanhf(c_new);
    Cst [b * Udim + uidx] = c_new;
    Hout[b * Udim + uidx] = h_new;
    const float fm = dropmask((uint32_t)(b * Ldim + t), (uint32_t)uidx, 3u);
    Hseq[(size_t)(b * Ldim + t) * Udim + uidx] = (__bf16)(h_new * fm);
  }
}

// ---------------------------------------------------------------------------
// Host-side orchestration
// ---------------------------------------------------------------------------
extern "C" void kernel_launch(void* const* d_in, const int* in_sizes, int n_in,
                              void* d_out, int out_size, void* d_ws, size_t ws_size,
                              hipStream_t stream) {
  // Input order: pos_embs(unused), images_emb, targets, emb_table,
  //              W_in, W_rec, b_lstm, W_out, b_out
  const float* img    = (const float*)d_in[1];
  const int*   tgt    = (const int*)  d_in[2];
  const float* emb    = (const float*)d_in[3];
  const float* W_in   = (const float*)d_in[4];
  const float* W_rec  = (const float*)d_in[5];
  const float* b_lstm = (const float*)d_in[6];
  const float* W_out  = (const float*)d_in[7];
  const float* b_out  = (const float*)d_in[8];
  float* out = (float*)d_out;

  // Carve workspace (256B aligned)
  char* ws = (char*)d_ws;
  auto carve = [&](size_t bytes) -> char* {
    char* p = ws;
    ws += (bytes + 255) & ~(size_t)255;
    return p;
  };
  __bf16* X     = (__bf16*)carve((size_t)Bdim * Ldim * Edim * 2);     // 2 MB
  __bf16* WinT  = (__bf16*)carve((size_t)FourU * Edim * 2);           // 2 MB
  __bf16* WrecT = (__bf16*)carve((size_t)FourU * Udim * 2);           // 2 MB
  __bf16* WoutT = (__bf16*)carve((size_t)Vdim * Udim * 2);            // 32.8 MB
  float*  Zin   = (float*) carve((size_t)Bdim * Ldim * FourU * 4);    // 16.8 MB
  __bf16* Hseq  = (__bf16*)carve((size_t)Bdim * Ldim * Udim * 2);     // 2 MB
  float*  H0    = (float*) carve((size_t)Bdim * Udim * 4);
  float*  H1    = (float*) carve((size_t)Bdim * Udim * 4);
  float*  Cst   = (float*) carve((size_t)Bdim * Udim * 4);

  // 1) Prep: embeddings+masks -> bf16 X; transpose+downconvert weights
  k_build_x<<<(Bdim * Ldim * Edim) / 256, 256, 0, stream>>>(img, tgt, emb, X);
  k_cvt_transpose<<<(Edim * FourU) / 256, 256, 0, stream>>>(W_in,  WinT,  Edim, FourU);
  k_cvt_transpose<<<(Edim * FourU) / 256, 256, 0, stream>>>(W_rec, WrecT, Udim, FourU);
  k_cvt_transpose<<<(Edim * Vdim)  / 256, 256, 0, stream>>>(W_out, WoutT, Udim, Vdim);
  k_init_state<<<(Bdim * Udim + 255) / 256, 256, 0, stream>>>(H0, Cst);

  // 2) Z_in = X @ W_in + b_lstm for all timesteps:  M=2048, N=2048, K=512
  {
    const int M = Bdim * Ldim, N = FourU, K = Edim;
    const int waves = (M / 16) * (N / 64);
    k_gemm_bf16<<<waves / 8, 256, 0, stream>>>(X, WinT, b_lstm, Zin, M, N, K);
  }

  // 3) Sequential recurrence: 64 launches, H ping-pong (stream-ordered)
  for (int t = 0; t < Ldim; ++t) {
    const float* Hin  = (t & 1) ? H1 : H0;
    float*       Hout = (t & 1) ? H0 : H1;
    k_lstm_step<<<Udim / 16, 256, 0, stream>>>(Zin, WrecT, Hin, Hout, Cst, Hseq, t);
  }

  // 4) predictions = Hseq @ W_out + b_out:  M=2048, N=32000, K=512
  {
    const int M = Bdim * Ldim, N = Vdim, K = Udim;
    const int waves = (M / 16) * (N / 64);
    k_gemm_bf16<<<waves / 8, 256, 0, stream>>>(Hseq, WoutT, b_out, out, M, N, K);
  }
}
